// Self_Attention_65197603553338
// MI455X (gfx1250) — matrix operand order repaired and checked by reading, compiled, not run on hardware
//
#include <hip/hip_runtime.h>
#include <hip/hip_bf16.h>

typedef __attribute__((ext_vector_type(16))) _Float16 v16h;
typedef __attribute__((ext_vector_type(8)))  _Float16 v8h;
typedef __attribute__((ext_vector_type(4)))  _Float16 v4h;
typedef __attribute__((ext_vector_type(8)))  float    v8f;
typedef __attribute__((ext_vector_type(4)))  float    v4f;

#define B_   256
#define S_   300
#define SP_  320   // sequence padded to 20 tiles of 16
#define D_   512
#define SCALE 0.05773502691896258f  // 1/sqrt(300)

// Build a 16-element f16 fragment from two 16B halves.
static __device__ __forceinline__ v16h mk16(v8h lo, v8h hi) {
  v16h r;
#pragma unroll
  for (int i = 0; i < 8; i++) { r[i] = lo[i]; r[i + 8] = hi[i]; }
  return r;
}

// Byte offset of a generic pointer within the wave's LDS allocation.
static __device__ __forceinline__ unsigned lds_off(void* p) {
  return (unsigned)(unsigned long long)(__attribute__((address_space(3))) char*)p;
}

// CDNA5 async global->LDS copy, 16B per lane, tracked by ASYNCcnt.
static __device__ __forceinline__ void async_b128(unsigned ldsoff, const void* g) {
  asm volatile("global_load_async_to_lds_b128 %0, %1, off"
               :
               : "v"(ldsoff), "v"(g)
               : "memory");
}
static __device__ __forceinline__ void wait_async0() {
  asm volatile("s_wait_asynccnt 0x0" ::: "memory");
}

// ---------------------------------------------------------------------------
// Kernel 1: W[3][k][n] f32  ->  WT[3][n][k] f16  (tiny: 3*512*512)
// ---------------------------------------------------------------------------
__global__ __launch_bounds__(256) void k_wt(const float* __restrict__ w,
                                            _Float16* __restrict__ wt) {
  int idx = blockIdx.x * 256 + threadIdx.x;
  int g = idx >> 18;
  int r = idx & 262143;
  int n = r >> 9, k = r & 511;
  wt[idx] = (_Float16)w[(g << 18) + (k << 9) + n];
}

// ---------------------------------------------------------------------------
// Kernel 2: QKV projection.  grid = (B*10, 1, 3), block = 256 (8 waves).
// Block: 32-row s-tile x 512 cols.  x-tile async-staged to LDS (f32), converted
// to f16 once, then each wave does 2 m-subtiles x 4 n-tiles (B reused 2x).
// Q,K -> row-major f16 [B][320][512]; V -> transposed f16 [B][512][320].
// Dynamic LDS: [0,64K) f32 stage / V-transpose scratch; [64K,96K) f16 A-tile.
// ---------------------------------------------------------------------------
__global__ __launch_bounds__(256) void k_qkv(const float* __restrict__ x,
                                             const _Float16* __restrict__ wt,
                                             _Float16* __restrict__ qf,
                                             _Float16* __restrict__ kf,
                                             _Float16* __restrict__ vt) {
  extern __shared__ __align__(16) char smem[];
  float*    sA32 = (float*)smem;                 // 32*512*4 = 64KB
  _Float16* sA16 = (_Float16*)(smem + 65536);    // 32*512*2 = 32KB
  _Float16* ldsv = (_Float16*)smem;              // reuse: 8 waves * 64n * 32s f16

  const int tid = threadIdx.x;
  const int mt = blockIdx.x;          // 0 .. B*10-1
  const int g  = blockIdx.z;          // 0=Q 1=K 2=V
  const int b  = mt / 10;
  const int sbase = (mt % 10) * 32;
  const int lane = tid & 31;
  const int w    = tid >> 5;
  const int kh   = lane >> 4;
  const int ln   = lane & 15;

  // ---- stage x tile (32 rows, clamped pad rows) into LDS via async copies ----
#pragma unroll
  for (int i = 0; i < 16; i++) {
    int flat = i * 256 + tid;          // 4096 chunks of 16B
    int row = flat >> 7, chunk = flat & 127;
    int srow = sbase + row;
    int sc = srow < S_ ? srow : (S_ - 1);
    const float* gp = x + (size_t)(b * S_ + sc) * D_ + chunk * 4;
    async_b128(lds_off(sA32 + row * 512 + chunk * 4), gp);
  }
  wait_async0();
  __syncthreads();

  // ---- convert staged tile to f16 once (each thread: 64 elements) ----
#pragma unroll
  for (int i = 0; i < 16; i++) {
    int idx = (i * 256 + tid) * 4;
    v4f f = *(const v4f*)(sA32 + idx);
    v4h h;
#pragma unroll
    for (int j = 0; j < 4; j++) h[j] = (_Float16)f[j];
    *(v4h*)(sA16 + idx) = h;
  }
  __syncthreads();

  const _Float16* wg = wt + ((size_t)g << 18);   // WT[g][n][k]

  v8f acc[2][4];
#pragma unroll
  for (int s2 = 0; s2 < 2; s2++)
#pragma unroll
    for (int t = 0; t < 4; t++) acc[s2][t] = {};

  for (int ks = 0; ks < D_; ks += 32) {
    // A fragments for both 16-row subtiles from LDS f16
    v16h a[2];
#pragma unroll
    for (int s2 = 0; s2 < 2; s2++) {
      const _Float16* ap = sA16 + (s2 * 16 + ln) * 512 + ks + kh * 8;
      a[s2] = mk16(*(const v8h*)ap, *(const v8h*)(ap + 16));
    }
#pragma unroll
    for (int t = 0; t < 4; t++) {
      int n = w * 64 + t * 16 + ln;
      const _Float16* bp = wg + ((size_t)n << 9) + ks + kh * 8;
      v16h bb = mk16(*(const v8h*)bp, *(const v8h*)(bp + 16));
      acc[0][t] = __builtin_amdgcn_wmma_f32_16x16x32_f16(
          false, a[0], false, bb, (short)0, acc[0][t], false, false);
      acc[1][t] = __builtin_amdgcn_wmma_f32_16x16x32_f16(
          false, a[1], false, bb, (short)0, acc[1][t], false, false);
    }
  }

  if (g < 2) {
    _Float16* out = (g == 0) ? qf : kf;
#pragma unroll
    for (int s2 = 0; s2 < 2; s2++)
#pragma unroll
      for (int t = 0; t < 4; t++) {
        int n = w * 64 + t * 16 + ln;
#pragma unroll
        for (int r = 0; r < 8; r++) {
          int s = sbase + s2 * 16 + r + kh * 8;
          out[(size_t)(b * SP_ + s) * D_ + n] = (_Float16)acc[s2][t][r];
        }
      }
  } else {
    // V: transpose through LDS; per-wave region [64 n][32 s]
    _Float16* lv = ldsv + w * (64 * 32);
#pragma unroll
    for (int s2 = 0; s2 < 2; s2++)
#pragma unroll
      for (int t = 0; t < 4; t++) {
        int nl = t * 16 + ln;
#pragma unroll
        for (int r = 0; r < 8; r++)
          lv[nl * 32 + s2 * 16 + r + kh * 8] = (_Float16)acc[s2][t][r];
      }
    __syncthreads();  // uniform: all blocks with g==2
#pragma unroll
    for (int uu = 0; uu < 8; uu++) {
      int c = uu * 32 + lane;            // 256 chunks of 16B per wave
      int nl = c >> 2, qq = c & 3;
      v8h val = *(const v8h*)(lv + nl * 32 + qq * 8);
      _Float16* dst = vt + (size_t)(b * D_ + w * 64 + nl) * SP_ + sbase + qq * 8;
      *(v8h*)dst = val;
    }
  }
}

// ---------------------------------------------------------------------------
// Kernel 3: attention.  grid = (19, B), block = 256 (8 waves).
// Q tile async-staged to LDS; scores -> softmax -> P@V.
// ---------------------------------------------------------------------------
__global__ __launch_bounds__(256) void k_attn(const _Float16* __restrict__ qf,
                                              const _Float16* __restrict__ kf,
                                              const _Float16* __restrict__ vt,
                                              float* __restrict__ out) {
  __shared__ __attribute__((aligned(16))) _Float16 sQ[16][D_];   // 16KB
  __shared__ __attribute__((aligned(16))) float    sS[16][SP_];  // 20KB
  __shared__ __attribute__((aligned(16))) _Float16 sP[16][SP_];  // 10KB

  const int tid = threadIdx.x;
  const int qt = blockIdx.x;
  const int b  = blockIdx.y;
  const int lane = tid & 31;
  const int w    = tid >> 5;
  const int kh = lane >> 4, ln = lane & 15;

  // ---- stage Q tile (16 x 512 f16) into LDS via async copies ----
#pragma unroll
  for (int i = 0; i < 4; i++) {
    int flat = i * 256 + tid;          // 1024 chunks of 16B
    int row = flat >> 6, chunk = flat & 63;
    const _Float16* gp = qf + (size_t)(b * SP_ + qt * 16 + row) * D_ + chunk * 8;
    async_b128(lds_off(&sQ[row][chunk * 8]), gp);
  }
  wait_async0();
  __syncthreads();

  // ---- phase 1: S = scale * Q Kt, all 16x320 into LDS ----
  for (int kt = w; kt < 20; kt += 8) {
    int kbase = kt * 16;
    const _Float16* krow = kf + (size_t)(b * SP_ + kbase + ln) * D_;
    v8f acc = {};
    for (int d = 0; d < D_; d += 32) {
      const _Float16* ap = &sQ[ln][d + kh * 8];
      v16h a = mk16(*(const v8h*)ap, *(const v8h*)(ap + 16));
      const _Float16* bp = krow + d + kh * 8;
      v16h bb = mk16(*(const v8h*)bp, *(const v8h*)(bp + 16));
      acc = __builtin_amdgcn_wmma_f32_16x16x32_f16(
          false, a, false, bb, (short)0, acc, false, false);
    }
#pragma unroll
    for (int r = 0; r < 8; r++)
      sS[r + kh * 8][kbase + ln] = acc[r] * SCALE;
  }
  __syncthreads();

  // ---- phase 2: softmax per row (row spread over 16 lanes, wave-local) ----
  {
    int q = 2 * w + kh;
    float p[20];
    float mx = -1e30f;
#pragma unroll
    for (int i = 0; i < 20; i++) {
      int col = ln + 16 * i;
      float s = sS[q][col];
      if (col >= S_) s = -1e30f;        // mask padded keys
      p[i] = s;
      mx = fmaxf(mx, s);
    }
#pragma unroll
    for (int m = 8; m >= 1; m >>= 1) mx = fmaxf(mx, __shfl_xor(mx, m, 32));
    float sum = 0.f;
#pragma unroll
    for (int i = 0; i < 20; i++) { p[i] = __expf(p[i] - mx); sum += p[i]; }
#pragma unroll
    for (int m = 8; m >= 1; m >>= 1) sum += __shfl_xor(sum, m, 32);
    float inv = 1.0f / sum;
#pragma unroll
    for (int i = 0; i < 20; i++)
      sP[q][ln + 16 * i] = (_Float16)(p[i] * inv);
  }
  __syncthreads();

  // ---- phase 3: O = P @ V (A from LDS, B from Vt) ----
  v8f o[4];
#pragma unroll
  for (int t = 0; t < 4; t++) o[t] = {};
  for (int kb = 0; kb < SP_; kb += 32) {
    const _Float16* ap = &sP[ln][kb + kh * 8];
    v16h a = mk16(*(const v8h*)ap, *(const v8h*)(ap + 16));
#pragma unroll
    for (int t = 0; t < 4; t++) {
      int n = w * 64 + t * 16 + ln;
      const _Float16* bp = vt + (size_t)(b * D_ + n) * SP_ + kb + kh * 8;
      v16h bb = mk16(*(const v8h*)bp, *(const v8h*)(bp + 16));
      o[t] = __builtin_amdgcn_wmma_f32_16x16x32_f16(
          false, a, false, bb, (short)0, o[t], false, false);
    }
  }

#pragma unroll
  for (int t = 0; t < 4; t++) {
    int n = w * 64 + t * 16 + ln;
#pragma unroll
    for (int r = 0; r < 8; r++) {
      int q = qt * 16 + r + kh * 8;
      if (q < S_)
        out[(size_t)(b * S_ + q) * D_ + n] = o[t][r];
    }
  }
}

// ---------------------------------------------------------------------------
extern "C" void kernel_launch(void* const* d_in, const int* in_sizes, int n_in,
                              void* d_out, int out_size, void* d_ws, size_t ws_size,
                              hipStream_t stream) {
  (void)in_sizes; (void)n_in; (void)out_size; (void)ws_size;
  const float* x  = (const float*)d_in[0];
  const float* wm = (const float*)d_in[1];
  float* out = (float*)d_out;

  char* ws = (char*)d_ws;
  _Float16* wt = (_Float16*)ws;                       // 3*512*512 f16
  size_t off = (size_t)3 * 512 * 512 * 2;
  _Float16* qf = (_Float16*)(ws + off); off += (size_t)B_ * SP_ * D_ * 2;
  _Float16* kf = (_Float16*)(ws + off); off += (size_t)B_ * SP_ * D_ * 2;
  _Float16* vt = (_Float16*)(ws + off);               // [B][512][320] f16

  k_wt  <<<3072, 256, 0, stream>>>(wm, wt);
  k_qkv <<<dim3(B_ * 10, 1, 3), 256, 98304, stream>>>(x, wt, qf, kf, vt);
  k_attn<<<dim3(19, B_), 256, 0, stream>>>(qf, kf, vt, out);
}
